// Model_84997402788288
// MI455X (gfx1250) — matrix operand order repaired and checked by reading, not compiled, not run on hardware
//
#include <hip/hip_runtime.h>
#include <math.h>

// ---------------- types for WMMA ----------------
typedef __attribute__((ext_vector_type(16))) __bf16       v16bf;
typedef __attribute__((ext_vector_type(8)))  float        v8f;
typedef __attribute__((ext_vector_type(4)))  unsigned int v4u;

union Frag { v4u u[2]; v16bf v; };

// ---------------- model constants ----------------
#define BB   512
#define PP   168
#define MM   321
#define MP   336              // padded M so K_conv % 32 == 0 and rows 16B aligned
#define HID  256
#define CKK  6
#define TT   163              // P - CK + 1
#define KC   (CKK * MP)       // 2016
#define HWN  24
#define XPAD 8192             // bf16 elements of tail padding for over-reads
#define NBLK 64               // persistent LSTM blocks (co-resident, device-wide sync)

__device__ __forceinline__ unsigned short f2bf(float f) {
  unsigned int u = __float_as_uint(f);
  u += 0x7fffu + ((u >> 16) & 1u);       // round-to-nearest-even
  return (unsigned short)(u >> 16);
}
__device__ __forceinline__ float bf2f(unsigned short s) {
  return __uint_as_float(((unsigned int)s) << 16);
}
__device__ __forceinline__ float sigm(float x) {
  return 1.0f / (1.0f + __expf(-x));
}

// A-fragment (16x32 bf16, MxK): lane half h holds K in {h*8+0..7, 16+h*8+0..7}
__device__ __forceinline__ v16bf load_afrag(const unsigned short* __restrict__ rowp, int half) {
  Frag f;
  const v4u* p = reinterpret_cast<const v4u*>(rowp + half * 8);
  f.u[0] = p[0];          // K = half*8 + 0..7
  f.u[1] = p[2];          // K = 16 + half*8 + 0..7   (+32 bytes)
  return f.v;
}
// B-fragment (32x16 bf16, KxN): lane half h holds K in {h*8+0..7, 16+h*8+0..7}
__device__ __forceinline__ v16bf load_bfrag(const unsigned short* __restrict__ rowp, int half) {
  Frag f;
  const v4u* p = reinterpret_cast<const v4u*>(rowp + half * 8);
  f.u[0] = p[0];          // K = half*8 + 0..7
  f.u[1] = p[2];          // K = 16 + half*8 + 0..7
  return f.v;
}
// B-fragment from LDS (same layout) -> ds_load_b128 x2
__device__ __forceinline__ v16bf load_bfrag_lds(const unsigned short* rowp, int half) {
  Frag f;
  const v4u* p = reinterpret_cast<const v4u*>(rowp + half * 8);
  f.u[0] = p[0];
  f.u[1] = p[2];
  return f.v;
}

// ---------------- pack / init kernels ----------------
__global__ void pack_x(const float* __restrict__ x, unsigned short* __restrict__ xbf, long n) {
  long i = (long)blockIdx.x * blockDim.x + threadIdx.x;
  if (i >= n) return;
  float v = 0.0f;
  if (i < (long)BB * PP * MP) {
    int mp = (int)(i % MP);
    long bp = i / MP;
    if (mp < MM) v = x[bp * MM + mp];
  }
  xbf[i] = f2bf(v);
}

__global__ void pack_convw(const float* __restrict__ w, unsigned short* __restrict__ cw) {
  int i = blockIdx.x * blockDim.x + threadIdx.x;
  if (i >= HID * KC) return;
  int hid = i / KC, kk = i % KC;
  int k = kk / MP, m = kk % MP;
  float v = (m < MM) ? w[(hid * CKK + k) * MM + m] : 0.0f;
  cw[i] = f2bf(v);
}

__global__ void pack_f32bf(const float* __restrict__ s, unsigned short* __restrict__ d, int n) {
  int i = blockIdx.x * blockDim.x + threadIdx.x;
  if (i < n) d[i] = f2bf(s[i]);
}

__global__ void init_state(const float* __restrict__ h0, const float* __restrict__ c0,
                           unsigned short* __restrict__ hsbf, float* __restrict__ cst,
                           unsigned int* __restrict__ counter) {
  int i = blockIdx.x * blockDim.x + threadIdx.x;
  if (i == 0) *counter = 0u;
  if (i >= BB * HID) return;
  hsbf[i] = f2bf(h0[i]);     // slot 0 of hs = h0
  cst[i]  = c0[i];
}

// ---------------- conv as GEMM: r[t,b,h] = relu(bias + patch(b,t) . w[h]) ----------------
// wave task: 16 t-rows x 64 h-cols (4 accumulators). 22528 waves = 2816 blocks.
__global__ void conv_gemm(const unsigned short* __restrict__ xbf,
                          const unsigned short* __restrict__ cw,
                          const float* __restrict__ conv_b,
                          unsigned short* __restrict__ rbf) {
  int wave = (blockIdx.x * blockDim.x + threadIdx.x) >> 5;
  int lane = threadIdx.x & 31;
  int half = lane >> 4, ln = lane & 15;
  int cg = wave & 3;                 // col group: 64 cols
  int rt = wave >> 2;                // 0..5631
  int b  = rt / 11;
  int t0 = (rt % 11) * 16;

  const unsigned short* arow = xbf + (long)(b * PP + t0 + ln) * MP;
  const unsigned short* wrow[4];
#pragma unroll
  for (int g = 0; g < 4; ++g)
    wrow[g] = cw + (long)(cg * 64 + g * 16 + ln) * KC;

  v8f acc[4];
#pragma unroll
  for (int g = 0; g < 4; ++g) acc[g] = (v8f){0,0,0,0,0,0,0,0};

  for (int kb = 0; kb < KC; kb += 32) {
    v16bf a = load_afrag(arow + kb, half);
#pragma unroll
    for (int g = 0; g < 4; ++g) {
      v16bf bm = load_bfrag(wrow[g] + kb, half);
      acc[g] = __builtin_amdgcn_wmma_f32_16x16x32_bf16(
          false, a, false, bm, (short)0, acc[g], false, false);
    }
  }
#pragma unroll
  for (int g = 0; g < 4; ++g) {
    int h = cg * 64 + g * 16 + ln;
    float bias = conv_b[h];
#pragma unroll
    for (int i = 0; i < 8; ++i) {
      int t = t0 + i + 8 * half;
      if (t < TT) {
        float v = acc[g][i] + bias;
        v = v > 0.0f ? v : 0.0f;
        rbf[((long)t * BB + b) * HID + h] = f2bf(v);
      }
    }
  }
}

// ---------------- persistent LSTM scan ----------------
// 64 blocks x 8 waves. Block = (bgroup 0..3, hcol 0..15): all 8 waves share one
// hid-tile, so the block's 64 weight rows per matrix live in LDS (64KB total).
// Device-wide release/acquire barrier between steps (blocks are co-resident).
__global__ void lstm_scan(const unsigned short* __restrict__ rbf,
                          unsigned short* __restrict__ hsbf,
                          const unsigned short* __restrict__ wih,
                          const unsigned short* __restrict__ whh,
                          const float* __restrict__ b_ih,
                          const float* __restrict__ b_hh,
                          float* __restrict__ cst,
                          unsigned int* __restrict__ counter) {
  __shared__ unsigned short lw[2][64 * HID];   // [ih/hh][(g*16+r)*256 + k]

  int bid = blockIdx.x;
  int tid = threadIdx.x;
  int hcol = bid & 15;
  int bgrp = bid >> 4;                         // 0..3
  int wv = tid >> 5, lane = tid & 31;
  int half = lane >> 4, ln = lane & 15;
  int hid0 = hcol * 16;
  int b0 = (bgrp * 8 + wv) * 16;

  // stage the 64 weight rows (j = g*256 + hid0 + r) of each matrix into LDS
  for (int e = tid; e < 64 * HID; e += 256) {
    int row = e >> 8;                          // 0..63
    int k = e & 255;
    long j = (long)((row >> 4) * 256 + hid0 + (row & 15));
    lw[0][e] = wih[j * HID + k];
    lw[1][e] = whh[j * HID + k];
  }
  __syncthreads();

  int j0 = hid0 + ln;
  float bi = b_ih[j0]       + b_hh[j0];
  float bf = b_ih[j0 + 256] + b_hh[j0 + 256];
  float bg = b_ih[j0 + 512] + b_hh[j0 + 512];
  float bo = b_ih[j0 + 768] + b_hh[j0 + 768];

  for (int t = 0; t < TT; ++t) {
    const unsigned short* xrow = rbf  + ((long)t * BB + b0 + ln) * HID;
    const unsigned short* hrow = hsbf + ((long)t * BB + b0 + ln) * HID;

    v8f acc[4];
#pragma unroll
    for (int g = 0; g < 4; ++g) acc[g] = (v8f){0,0,0,0,0,0,0,0};

    for (int kb = 0; kb < HID; kb += 32) {     // x-part (A: global, B: LDS)
      v16bf a = load_afrag(xrow + kb, half);
#pragma unroll
      for (int g = 0; g < 4; ++g) {
        v16bf bm = load_bfrag_lds(&lw[0][(g * 16 + ln) * HID + kb], half);
        acc[g] = __builtin_amdgcn_wmma_f32_16x16x32_bf16(
            false, a, false, bm, (short)0, acc[g], false, false);
      }
    }
    for (int kb = 0; kb < HID; kb += 32) {     // h-part
      v16bf a = load_afrag(hrow + kb, half);
#pragma unroll
      for (int g = 0; g < 4; ++g) {
        v16bf bm = load_bfrag_lds(&lw[1][(g * 16 + ln) * HID + kb], half);
        acc[g] = __builtin_amdgcn_wmma_f32_16x16x32_bf16(
            false, a, false, bm, (short)0, acc[g], false, false);
      }
    }

#pragma unroll
    for (int i = 0; i < 8; ++i) {
      int b = b0 + i + 8 * half;
      long ci = (long)b * HID + hid0 + ln;
      float iv = sigm(acc[0][i] + bi);
      float fv = sigm(acc[1][i] + bf);
      float gv = tanhf(acc[2][i] + bg);
      float ov = sigm(acc[3][i] + bo);
      float cn = fv * cst[ci] + iv * gv;
      cst[ci] = cn;
      hsbf[((long)(t + 1) * BB) * HID + ci] = f2bf(ov * tanhf(cn));
    }

    // device-wide barrier: release our h-writes, acquire everyone else's
    __syncthreads();
    if (tid == 0) {
      __hip_atomic_fetch_add(counter, 1u, __ATOMIC_RELEASE, __HIP_MEMORY_SCOPE_AGENT);
      unsigned int target = (unsigned int)NBLK * (unsigned int)(t + 1);
      while (__hip_atomic_load(counter, __ATOMIC_ACQUIRE, __HIP_MEMORY_SCOPE_AGENT) < target) {
        __builtin_amdgcn_s_sleep(2);
      }
    }
    __syncthreads();
  }
}

// ---------------- attention pooling: one block per batch row ----------------
__global__ void attn_pool(const unsigned short* __restrict__ hsbf,
                          const float* __restrict__ att_w,
                          const float* __restrict__ att_b,
                          float* __restrict__ pooled) {
  __shared__ float ssc[176];
  __shared__ float sred;
  int b = blockIdx.x, tid = threadIdx.x;
  if (tid < TT) {
    const unsigned short* h = hsbf + ((long)(tid + 1) * BB + b) * HID;
    float a = 0.0f;
    for (int k = 0; k < HID; ++k) a += bf2f(h[k]) * att_w[k];
    ssc[tid] = a + att_b[0];
  }
  __syncthreads();
  if (tid == 0) {
    float mx = -1e30f;
    for (int t = 0; t < TT; ++t) mx = fmaxf(mx, ssc[t]);
    sred = mx;
  }
  __syncthreads();
  if (tid < TT) ssc[tid] = __expf(ssc[tid] - sred);
  __syncthreads();
  if (tid == 0) {
    float s = 0.0f;
    for (int t = 0; t < TT; ++t) s += ssc[t];
    sred = 1.0f / s;
  }
  __syncthreads();
  if (tid < TT) ssc[tid] *= sred;
  __syncthreads();
  float acc = 0.0f;                               // tid == hid channel
  for (int t = 0; t < TT; ++t)
    acc += ssc[t] * bf2f(hsbf[((long)(t + 1) * BB + b) * HID + tid]);
  pooled[(long)b * HID + tid] = acc;
}

// ---------------- final linear + highway ----------------
__global__ void final_out(const float* __restrict__ pooled,
                          const float* __restrict__ lin_w,
                          const float* __restrict__ lin_b,
                          const float* __restrict__ x,
                          const float* __restrict__ hw_w,
                          const float* __restrict__ hw_b,
                          float* __restrict__ out) {
  int i = blockIdx.x * blockDim.x + threadIdx.x;
  if (i >= BB * MM) return;
  int b = i / MM, m = i % MM;
  float acc = lin_b[m];
  for (int k = 0; k < HID; ++k) acc += pooled[(long)b * HID + k] * lin_w[(long)m * HID + k];
  float z = hw_b[0];
#pragma unroll
  for (int j = 0; j < HWN; ++j)
    z += x[((long)b * PP + (PP - HWN) + j) * MM + m] * hw_w[j];
  out[i] = acc + z;
}

// ---------------- launch ----------------
extern "C" void kernel_launch(void* const* d_in, const int* in_sizes, int n_in,
                              void* d_out, int out_size, void* d_ws, size_t ws_size,
                              hipStream_t stream) {
  const float* x      = (const float*)d_in[0];
  const float* h0     = (const float*)d_in[1];
  const float* c0     = (const float*)d_in[2];
  const float* conv_w = (const float*)d_in[3];
  const float* conv_b = (const float*)d_in[4];
  const float* W_ih   = (const float*)d_in[5];
  const float* W_hh   = (const float*)d_in[6];
  const float* b_ih   = (const float*)d_in[7];
  const float* b_hh   = (const float*)d_in[8];
  const float* att_w  = (const float*)d_in[9];
  const float* att_b  = (const float*)d_in[10];
  const float* lin_w  = (const float*)d_in[11];
  const float* lin_b  = (const float*)d_in[12];
  const float* hw_w   = (const float*)d_in[13];
  const float* hw_b   = (const float*)d_in[14];
  float* out = (float*)d_out;

  char* base = (char*)d_ws;
  size_t off = 0;
  auto carve = [&](size_t bytes) -> void* {
    void* p = base + off;
    off += (bytes + 255) & ~(size_t)255;
    return p;
  };
  const long NX = (long)BB * PP * MP + XPAD;
  unsigned short* xbf  = (unsigned short*)carve((size_t)NX * 2);
  unsigned short* cw   = (unsigned short*)carve((size_t)HID * KC * 2);
  unsigned short* rbf  = (unsigned short*)carve((size_t)TT * BB * HID * 2);
  unsigned short* wih  = (unsigned short*)carve((size_t)4 * HID * HID * 2);
  unsigned short* whh  = (unsigned short*)carve((size_t)4 * HID * HID * 2);
  unsigned short* hsbf = (unsigned short*)carve((size_t)(TT + 1) * BB * HID * 2);
  float*          cst  = (float*)carve((size_t)BB * HID * 4);
  float*          pld  = (float*)carve((size_t)BB * HID * 4);
  unsigned int*   ctr  = (unsigned int*)carve(256);
  (void)ws_size; (void)n_in; (void)in_sizes; (void)out_size;

  // pack / init
  pack_x<<<(unsigned)((NX + 255) / 256), 256, 0, stream>>>(x, xbf, NX);
  pack_convw<<<(HID * KC + 255) / 256, 256, 0, stream>>>(conv_w, cw);
  pack_f32bf<<<(4 * HID * HID + 255) / 256, 256, 0, stream>>>(W_ih, wih, 4 * HID * HID);
  pack_f32bf<<<(4 * HID * HID + 255) / 256, 256, 0, stream>>>(W_hh, whh, 4 * HID * HID);
  init_state<<<(BB * HID + 255) / 256, 256, 0, stream>>>(h0, c0, hsbf, cst, ctr);

  // conv GEMM: 512 b * 11 row-tiles * 4 col-groups = 22528 waves / 8 per block
  conv_gemm<<<2816, 256, 0, stream>>>(xbf, cw, conv_b, rbf);

  // persistent LSTM scan: one kernel, internal device-wide sync per step
  lstm_scan<<<NBLK, 256, 0, stream>>>(rbf, hsbf, wih, whh, b_ih, b_hh, cst, ctr);

  // attention pooling + output
  attn_pool<<<BB, 256, 0, stream>>>(hsbf, att_w, att_b, pld);
  final_out<<<(BB * MM + 255) / 256, 256, 0, stream>>>(pld, lin_w, lin_b, x, hw_w, hw_b, out);
}
